// ProxyDecoderLoss_7593502179727
// MI455X (gfx1250) — matrix operand order repaired and checked by reading, not compiled, not run on hardware
//
#include <hip/hip_runtime.h>
#include <hip/hip_bf16.h>

typedef __attribute__((ext_vector_type(16))) _Float16 v16h;
typedef __attribute__((ext_vector_type(8)))  _Float16 v8h;
typedef __attribute__((ext_vector_type(2)))  __fp16   v2hp;   // cvt_pkrtz result type
typedef __attribute__((ext_vector_type(8)))  float    v8f;
typedef __attribute__((ext_vector_type(2)))  float    v2f;

#define BB 8
#define PP 64
#define GG 32
#define NN 100000
#define WAVES 8
#define SLICES 25
#define ASTRIDE 40  // halfs per A row in LDS (80B) -> conflict-free b128 access

// ---------------------------------------------------------------------------
// Kernel 0: zero the accumulation workspace (num + cnt)
// ---------------------------------------------------------------------------
__global__ void zero_kernel(float* __restrict__ p, int n) {
    int i = blockIdx.x * blockDim.x + threadIdx.x;
    if (i < n) p[i] = 0.0f;
}

// ---------------------------------------------------------------------------
// Kernel 1: num[b,p,g] = sum_n |points[b,n]·nrm[b,p] - d[b,p]| * mask[b,g,n]
//           cnt[b,g]   = sum_n mask[b,g,n]
// proj generated by f32 WMMA (K=4: x,y,z,1 vs nx,ny,nz,-d), accumulated by
// f16 WMMA (K=32).  grid = (B, SLICES), block = 256 (8 waves)
// ---------------------------------------------------------------------------
__global__ __launch_bounds__(256)
void chamfer_wmma_kernel(const float* __restrict__ pred_normals,
                         const float* __restrict__ pred_distances,
                         const unsigned char* __restrict__ gt_masks,
                         const float* __restrict__ points,
                         float* __restrict__ num_ws,
                         float* __restrict__ cnt_ws) {
    __shared__ float accs[PP * GG];              // workgroup-level f32 reduction
    __shared__ __align__(16) _Float16 projA[WAVES][PP * ASTRIDE];

    const int b    = blockIdx.x;
    const int s    = blockIdx.y;
    const int tid  = threadIdx.x;
    const int w    = tid >> 5;
    const int lane = tid & 31;
    const int lm   = lane & 15;
    const int hs   = lane >> 4;                  // half-wave select

    for (int i = tid; i < PP * GG; i += 256) accs[i] = 0.0f;
    __syncthreads();

    const v8f zero8 = {0.f, 0.f, 0.f, 0.f, 0.f, 0.f, 0.f, 0.f};
    v8f acc[4][2];
#pragma unroll
    for (int mi = 0; mi < 4; ++mi)
#pragma unroll
        for (int gi = 0; gi < 2; ++gi) acc[mi][gi] = zero8;

    // ---- loop-invariant B fragments of the small f32 GEMM (predictions) ----
    // f32 B 4x16 layout: lanes 0-15 hold K0,K1 = (nx,ny); lanes 16-31 K2,K3 = (nz,-d)
    v2f bsm[4];
#pragma unroll
    for (int pi = 0; pi < 4; ++pi) {
        const int p = pi * 16 + lm;
        const float* pn = pred_normals + ((size_t)b * PP + p) * 3;
        float bx = pn[hs ? 2 : 0];
        float by = hs ? -pred_distances[b * PP + p] : pn[1];
        v2f t = {bx, by};
        bsm[pi] = t;
    }

    float cntAcc0 = 0.0f, cntAcc1 = 0.0f;

    const int CH = NN / 32;                      // 3125 chunks of 32 points
    const int gw = s * WAVES + w;
    const int W  = SLICES * WAVES;
    _Float16* myA = &projA[w][0];
    const float* ptsB = points + (size_t)b * NN * 3;
    const unsigned char* maskB = gt_masks + (size_t)b * GG * NN;

    for (int c = gw; c < CH; c += W) {
        const int k0 = c * 32;

        // ---- proj tiles via f32 WMMA, pack |.| to f16, transpose via LDS ----
#pragma unroll
        for (int kt = 0; kt < 2; ++kt) {
            // f32 A 16x4 layout: lanes 0-15 hold K0,K1 = (px,py);
            //                    lanes 16-31 hold K2,K3 = (pz,1.0)
            const float* pp = ptsB + (size_t)(k0 + kt * 16 + lm) * 3;
            float ax = pp[hs ? 2 : 0];
            float ay = hs ? 1.0f : pp[1];
            v2f afr = {ax, ay};
#pragma unroll
            for (int pi = 0; pi < 4; ++pi) {
                v8f d = __builtin_amdgcn_wmma_f32_16x16x4_f32(
                    false, afr, false, bsm[pi], (short)0, zero8, false, false);
                // lane holds pred = pi*16+lm, points kt*16 + 8*hs + r (r=0..7)
                union { v2hp h2[4]; v8h v8; } pk;
#pragma unroll
                for (int i = 0; i < 4; ++i)
                    pk.h2[i] = __builtin_amdgcn_cvt_pkrtz(fabsf(d[2 * i]),
                                                          fabsf(d[2 * i + 1]));
                *(v8h*)(myA + (pi * 16 + lm) * ASTRIDE + kt * 16 + 8 * hs) = pk.v8;
            }
        }
        asm volatile("s_wait_dscnt 0" ::: "memory");  // cross-lane LDS handoff

        // ---- B fragments of the big GEMM from global mask bytes ----
        v16h bf[2];
#pragma unroll
        for (int gi = 0; gi < 2; ++gi) {
            const int g = gi * 16 + lm;
            // B in contiguous k order, matching A: lanes 0-15 hold K 0..15, lanes 16-31 hold K 16..31
            const unsigned char* mrow = maskB + (size_t)g * NN + k0 + hs * 16;
            unsigned long long q0 = *(const unsigned long long*)(mrow);
            unsigned long long q1 = *(const unsigned long long*)(mrow + 8);
            float pc = (float)(__popcll(q0) + __popcll(q1));   // bytes are 0/1
            if (gi == 0) cntAcc0 += pc; else cntAcc1 += pc;
            // bytes(0/1) -> packed f16 pairs: v_perm + mul by 0x3C00
            union { v16h v; unsigned int u32[8]; } ub;
            unsigned int dw[4] = { (unsigned int)q0, (unsigned int)(q0 >> 32),
                                   (unsigned int)q1, (unsigned int)(q1 >> 32) };
#pragma unroll
            for (int i = 0; i < 4; ++i) {
                ub.u32[2 * i]     = __builtin_amdgcn_perm(0u, dw[i], 0x04010400u) * 0x3C00u;
                ub.u32[2 * i + 1] = __builtin_amdgcn_perm(0u, dw[i], 0x04030402u) * 0x3C00u;
            }
            bf[gi] = ub.v;
        }

        // ---- A fragments from LDS + 8 f16 WMMAs (4 M-tiles x 2 G-tiles) ----
#pragma unroll
        for (int mi = 0; mi < 4; ++mi) {
            const int m = mi * 16 + lm;
            const _Float16* row = myA + m * ASTRIDE;
            // lanes 0-15: halves = K{0..15}; lanes 16-31: K{16..31}
            v8h lo = *(const v8h*)(row + hs * 16);
            v8h hi = *(const v8h*)(row + hs * 16 + 8);
            v16h afrag = __builtin_shufflevector(lo, hi, 0, 1, 2, 3, 4, 5, 6, 7,
                                                 8, 9, 10, 11, 12, 13, 14, 15);
            acc[mi][0] = __builtin_amdgcn_wmma_f32_16x16x32_f16(
                false, afrag, false, bf[0], (short)0, acc[mi][0], false, false);
            acc[mi][1] = __builtin_amdgcn_wmma_f32_16x16x32_f16(
                false, afrag, false, bf[1], (short)0, acc[mi][1], false, false);
        }
    }

    // ---- reduce wave accumulators into workgroup LDS, then global ----
#pragma unroll
    for (int mi = 0; mi < 4; ++mi)
#pragma unroll
        for (int gi = 0; gi < 2; ++gi)
#pragma unroll
            for (int r = 0; r < 8; ++r) {
                const int m = mi * 16 + r + hs * 8;  // C layout: VGPR r -> M=r / r+8
                const int g = gi * 16 + lm;
                atomicAdd(&accs[m * GG + g], acc[mi][gi][r]);
            }
    atomicAdd(&cnt_ws[b * GG + lm],      cntAcc0);
    atomicAdd(&cnt_ws[b * GG + 16 + lm], cntAcc1);
    __syncthreads();
    for (int i = tid; i < PP * GG; i += 256)
        atomicAdd(&num_ws[b * PP * GG + i], accs[i]);
}

// ---------------------------------------------------------------------------
// Kernel 2: padded 64x64 double cost matrices for the Hungarian solver
// ---------------------------------------------------------------------------
__global__ void cost_kernel(const float* __restrict__ pred_normals,
                            const float* __restrict__ pred_distances,
                            const float* __restrict__ gt_normals,
                            const float* __restrict__ gt_distances,
                            const float* __restrict__ num_ws,
                            const float* __restrict__ cnt_ws,
                            double* __restrict__ cost_sq) {
    int idx = blockIdx.x * blockDim.x + threadIdx.x;
    if (idx >= BB * PP * 64) return;
    int j = idx & 63, p = (idx >> 6) & 63, b = idx >> 12;
    double cst = 0.0;
    if (j < GG) {
        const float* pn = pred_normals + ((size_t)b * PP + p) * 3;
        const float* gn = gt_normals + ((size_t)b * GG + j) * 3;
        float dotv = pn[0] * gn[0] + pn[1] * gn[1] + pn[2] * gn[2];
        float ns = 1.0f - fabsf(dotv);
        float dd = fabsf(pred_distances[b * PP + p] - gt_distances[b * GG + j]);
        float cnt = cnt_ws[b * GG + j];
        float num = num_ws[(b * PP + p) * GG + j];
        float cham = (cnt > 0.0f) ? num / fmaxf(cnt, 1.0f) : 1.0f;  // EMPTY_CHAMFER
        cst = (double)(ns + 0.5f * dd + 5.0f * cham);
    }
    cost_sq[idx] = cst;
}

// ---------------------------------------------------------------------------
// Kernel 3: Jonker-Volgenant assignment, one 64-thread block per batch
// ---------------------------------------------------------------------------
__global__ __launch_bounds__(64)
void hungarian_kernel(const double* __restrict__ cost_sq, int* __restrict__ colOf) {
    const int n = 64;
    const double INF = 1e18;
    const int b = blockIdx.x;
    const int t = threadIdx.x;               // handles column j = t+1
    __shared__ double u[n + 1], vv[n + 1], minv[n + 1];
    __shared__ int p[n + 1], way[n + 1];
    __shared__ unsigned char used[n + 1];
    __shared__ int j0_sh, j1_sh;
    __shared__ double redV[64];
    __shared__ int redI[64];
    const double* C = cost_sq + (size_t)b * 64 * 64;

    u[t + 1] = 0.0; vv[t + 1] = 0.0; p[t + 1] = 0; way[t + 1] = 0;
    if (t == 0) { u[0] = 0.0; vv[0] = 0.0; p[0] = 0; way[0] = 0; }
    __syncthreads();

    for (int i = 1; i <= n; ++i) {
        minv[t + 1] = INF; used[t + 1] = 0;
        if (t == 0) { p[0] = i; used[0] = 0; j0_sh = 0; }
        __syncthreads();
        while (true) {
            const int j0 = j0_sh;
            if (t == 0) used[j0] = 1;
            __syncthreads();
            const int i0 = p[j0];
            const int j = t + 1;
            double cand = INF;
            if (!used[j]) {
                double cur = C[(i0 - 1) * 64 + (j - 1)] - u[i0] - vv[j];
                if (cur < minv[j]) { minv[j] = cur; way[j] = j0; }
                cand = minv[j];
            }
            redV[t] = cand; redI[t] = j;
            __syncthreads();
            for (int off = 32; off >= 1; off >>= 1) {
                if (t < off) {
                    double v2 = redV[t + off]; int i2 = redI[t + off];
                    if (v2 < redV[t] || (v2 == redV[t] && i2 < redI[t])) {
                        redV[t] = v2; redI[t] = i2;
                    }
                }
                __syncthreads();
            }
            const double delta = redV[0];
            const int j1 = redI[0];
            if (used[j]) { u[p[j]] += delta; vv[j] -= delta; }
            else         { minv[j] -= delta; }
            if (t == 0)  { u[p[0]] += delta; vv[0] -= delta; j0_sh = j1; }
            __syncthreads();
            if (p[j1] == 0) { if (t == 0) j1_sh = j1; break; }
        }
        __syncthreads();
        if (t == 0) {  // augment
            int j0 = j1_sh;
            while (j0) { int jn = way[j0]; p[j0] = p[jn]; j0 = jn; }
        }
        __syncthreads();
    }
    colOf[b * 64 + (p[t + 1] - 1)] = t;  // row_to_col
}

// ---------------------------------------------------------------------------
// Kernel 4: final losses -> d_out = {total, cls, param, chamfer}
// ---------------------------------------------------------------------------
__global__ __launch_bounds__(256)
void loss_kernel(const float* __restrict__ pred_logits,
                 const float* __restrict__ pred_normals,
                 const float* __restrict__ pred_distances,
                 const float* __restrict__ gt_normals,
                 const float* __restrict__ gt_distances,
                 const float* __restrict__ num_ws,
                 const float* __restrict__ cnt_ws,
                 const int* __restrict__ colOf,
                 float* __restrict__ out) {
    __shared__ float clsS[256];
    __shared__ float parS[BB], chamS[BB];
    __shared__ int kS[BB];
    const int t = threadIdx.x;
    if (t < BB) { parS[t] = 0.0f; chamS[t] = 0.0f; kS[t] = 0; }
    __syncthreads();

    float cls = 0.0f;
    for (int idx = t; idx < BB * PP; idx += 256) {
        const int b = idx >> 6;
        const int col = colOf[idx];
        const float tgt = (col < GG) ? 1.0f : 0.0f;
        const float l = pred_logits[idx];
        const float sp = fmaxf(l, 0.0f) + log1pf(expf(-fabsf(l)));  // softplus
        cls += sp - tgt * l;
        if (col < GG) {
            const int g = col;
            const float* pn = pred_normals + (size_t)idx * 3;
            const float* gn = gt_normals + ((size_t)b * GG + g) * 3;
            float dotv = pn[0] * gn[0] + pn[1] * gn[1] + pn[2] * gn[2];
            float nm = 1.0f - fabsf(dotv);
            float dm = fabsf(pred_distances[idx] - gt_distances[b * GG + g]);
            float cnt = cnt_ws[b * GG + g];
            float cham = (cnt > 0.0f) ? num_ws[(size_t)idx * GG + g] / fmaxf(cnt, 1.0f)
                                      : 0.0f;
            atomicAdd(&parS[b], nm + dm);
            atomicAdd(&chamS[b], cham);
            atomicAdd(&kS[b], 1);
        }
    }
    clsS[t] = cls;
    __syncthreads();
    for (int off = 128; off >= 1; off >>= 1) {
        if (t < off) clsS[t] += clsS[t + off];
        __syncthreads();
    }
    if (t == 0) {
        const float cls_loss = clsS[0] / (float)(BB * PP);
        float par = 0.0f, cham = 0.0f;
        for (int b = 0; b < BB; ++b) {
            const float k = (float)max(kS[b], 1);
            par  += parS[b] / k;
            cham += chamS[b] / k;
        }
        const float param_loss   = par / (float)BB;
        const float chamfer_loss = cham / (float)BB;
        out[0] = cls_loss + 0.5f * param_loss + 20.0f * chamfer_loss;
        out[1] = cls_loss;
        out[2] = param_loss;
        out[3] = chamfer_loss;
    }
}

// ---------------------------------------------------------------------------
extern "C" void kernel_launch(void* const* d_in, const int* in_sizes, int n_in,
                              void* d_out, int out_size, void* d_ws, size_t ws_size,
                              hipStream_t stream) {
    (void)in_sizes; (void)n_in; (void)out_size; (void)ws_size;
    const float* pred_logits    = (const float*)d_in[0];
    const float* pred_normals   = (const float*)d_in[1];
    const float* pred_distances = (const float*)d_in[2];
    const float* gt_normals     = (const float*)d_in[3];
    const float* gt_distances   = (const float*)d_in[4];
    const unsigned char* gt_masks = (const unsigned char*)d_in[5];  // bool -> u8
    const float* points         = (const float*)d_in[6];
    float* out = (float*)d_out;

    // workspace layout
    float*  num_ws  = (float*)d_ws;                              // B*P*G f32
    float*  cnt_ws  = num_ws + BB * PP * GG;                     // B*G   f32
    double* cost_sq = (double*)((char*)d_ws + (BB * PP * GG + BB * GG) * 4);
    int*    colOf   = (int*)((char*)cost_sq + (size_t)BB * 64 * 64 * 8);

    const int nzero = BB * PP * GG + BB * GG;
    zero_kernel<<<(nzero + 255) / 256, 256, 0, stream>>>(num_ws, nzero);

    dim3 grid1(BB, SLICES);
    chamfer_wmma_kernel<<<grid1, 256, 0, stream>>>(
        pred_normals, pred_distances, gt_masks, points, num_ws, cnt_ws);

    const int ncost = BB * PP * 64;
    cost_kernel<<<(ncost + 255) / 256, 256, 0, stream>>>(
        pred_normals, pred_distances, gt_normals, gt_distances,
        num_ws, cnt_ws, cost_sq);

    hungarian_kernel<<<BB, 64, 0, stream>>>(cost_sq, colOf);

    loss_kernel<<<1, 256, 0, stream>>>(
        pred_logits, pred_normals, pred_distances, gt_normals, gt_distances,
        num_ws, cnt_ws, colOf, out);
}
